// ScalarGateRNN_48533130444804
// MI455X (gfx1250) — matrix operand order repaired and checked by reading, not compiled, not run on hardware
//
#include <hip/hip_runtime.h>
#include <hip/hip_bf16.h>

#define B_  64
#define T_  2048
#define NI_ 128
#define NH_ 256
#define NO_ 64
#define KP  264   // padded bf16 row stride (elements): breaks LDS bank conflicts, keeps 16B alignment

typedef __attribute__((ext_vector_type(16))) __bf16 v16bf;
typedef __attribute__((ext_vector_type(8)))  __bf16 v8bf;
typedef __attribute__((ext_vector_type(8)))  float  v8f;

static __device__ __forceinline__ v16bf cat8(v8bf lo, v8bf hi) {
  return __builtin_shufflevector(lo, hi, 0,1,2,3,4,5,6,7,8,9,10,11,12,13,14,15);
}

static __device__ __forceinline__ float fast_tanh(float x) {
#if __has_builtin(__builtin_amdgcn_tanhf)
  return __builtin_amdgcn_tanhf(x);   // gfx1250 TANH trans op
#else
  return tanhf(x);
#endif
}

// ---------------------------------------------------------------------------
// Kernel 1: ui = u @ Wi^T, written straight into the As output region.
// u flat: [B*T, NI] row-major; As flat: [B*T, NH] row-major (same m index!).
// One 16x16 output tile per wave; K=128 in 4 bf16 WMMA steps, 2 accumulators.
// ---------------------------------------------------------------------------
__global__ void __launch_bounds__(256)
proj_ui_kernel(const float* __restrict__ u, const float* __restrict__ Wi,
               float* __restrict__ uiAs) {
  const int lane = threadIdx.x & 31;
  const int wv   = threadIdx.x >> 5;
  const int hl   = lane >> 4;      // half-wave select
  const int ln   = lane & 15;
  const int tile = blockIdx.x * 8 + wv;      // 131072 tiles total
  const int mt   = tile >> 4;                 // 8192 M tiles
  const int nb   = (tile & 15) << 4;          // 16 N tiles
  const long mrow = (long)mt * 16 + ln;       // A-matrix row for this lane

  v8f acc0 = {0.f,0.f,0.f,0.f,0.f,0.f,0.f,0.f};
  v8f acc1 = {0.f,0.f,0.f,0.f,0.f,0.f,0.f,0.f};
  const float* urow = u + mrow * NI_;
  #pragma unroll
  for (int kb = 0; kb < NI_; kb += 32) {
    // A (16x32 bf16): lane holds row ln, K = kb+16*hl+e (e<8) and kb+16*hl+8+e
    const float* ap0 = urow + kb + 16*hl;
    const float* ap1 = urow + kb + 16*hl + 8;
    // B (32x16 bf16): lane holds col ln, K = kb+16*hl+e, B[k][n] = Wi[n][k]
    const float* bp  = Wi + (long)(nb + ln) * NI_ + kb + 16*hl;
    v16bf a, b;
    #pragma unroll
    for (int e = 0; e < 8; ++e) { a[e] = (__bf16)ap0[e]; a[e+8] = (__bf16)ap1[e]; }
    #pragma unroll
    for (int e = 0; e < 16; ++e) b[e] = (__bf16)bp[e];
    if ((kb & 32) == 0)
      acc0 = __builtin_amdgcn_wmma_f32_16x16x32_bf16(false, a, false, b,
                                                     (short)0, acc0, false, false);
    else
      acc1 = __builtin_amdgcn_wmma_f32_16x16x32_bf16(false, a, false, b,
                                                     (short)0, acc1, false, false);
  }
  #pragma unroll
  for (int v = 0; v < 8; ++v) {
    long m = (long)mt * 16 + v + 8*hl;        // C layout: M = v + 8*(lane>>4)
    uiAs[m * NH_ + nb + ln] = acc0[v] + acc1[v];   // RT stores: scan re-reads soon
  }
}

// ---------------------------------------------------------------------------
// Kernel 2: ug = u @ Wig^T into the AGs output region. One wave per row.
// ---------------------------------------------------------------------------
__global__ void __launch_bounds__(256)
proj_ug_kernel(const float* __restrict__ u, const float* __restrict__ Wig,
               float* __restrict__ ugAG) {
  const int lane = threadIdx.x & 31;
  const int wv   = threadIdx.x >> 5;
  const long row = (long)blockIdx.x * 8 + wv;
  float p = 0.f;
  #pragma unroll
  for (int j = 0; j < 4; ++j) {
    int kk = lane + 32*j;
    p += u[row * NI_ + kk] * Wig[kk];
  }
  #pragma unroll
  for (int off = 16; off; off >>= 1) p += __shfl_xor(p, off, 32);
  if (lane == 0) ugAG[row] = p;
}

// ---------------------------------------------------------------------------
// Kernel 3: persistent scan. 4 workgroups x 512 threads; each WG owns 16
// independent batch rows (the recurrence is batch-separable). Wr lives in LDS
// as bf16 [256][KP]; x state lives in fp32 registers (8 elems/lane) with a
// bf16 LDS mirror used as the WMMA A-operand.
// ---------------------------------------------------------------------------
__global__ void __launch_bounds__(512)
scan_kernel(const float* __restrict__ Wr, const float* __restrict__ Wrg,
            float* __restrict__ Xo, float* __restrict__ Aso,
            float* __restrict__ Phio, float* __restrict__ Gso,
            float* __restrict__ AGo) {
  extern __shared__ __align__(16) char smem[];
  __bf16* sWr  = (__bf16*)smem;                                   // [NH_][KP]
  __bf16* sXb  = (__bf16*)(smem + (size_t)NH_*KP*2);              // [16][KP]
  float*  sWrg = (float*)(smem + (size_t)NH_*KP*2 + (size_t)16*KP*2);
  float*  sG   = sWrg + NH_;                                      // [16]

  const int tid  = threadIdx.x;
  const int lane = tid & 31, wv = tid >> 5;
  const int hl   = lane >> 4, ln = lane & 15;
  const int nb   = wv << 4;                 // this wave's 16 output columns
  const int bm   = blockIdx.x << 4;         // this WG's 16 batch rows

  // Stage Wr (bf16; B-operand wants Wr row-major) + Wrg; zero the x mirror.
  for (int idx = tid; idx < NH_*NH_; idx += 512) {
    int n = idx >> 8, k = idx & (NH_-1);
    sWr[n*KP + k] = (__bf16)Wr[idx];
  }
  for (int idx = tid; idx < NH_; idx += 512) sWrg[idx] = Wrg[idx];
  for (int idx = tid; idx < 16*KP; idx += 512) sXb[idx] = (__bf16)0.f;
  __syncthreads();

  float xold[8];
  #pragma unroll
  for (int v = 0; v < 8; ++v) xold[v] = 0.f;

  for (int t = 0; t < T_; ++t) {
    // --- recurrent GEMM tile: a_tile = x @ Wr^T (cols nb..nb+15), 2 accums
    v8f acc0 = {0.f,0.f,0.f,0.f,0.f,0.f,0.f,0.f};
    v8f acc1 = {0.f,0.f,0.f,0.f,0.f,0.f,0.f,0.f};
    #pragma unroll
    for (int kb = 0; kb < NH_; kb += 32) {
      v8bf alo = *(const v8bf*)&sXb[ln*KP + kb + 16*hl];
      v8bf ahi = *(const v8bf*)&sXb[ln*KP + kb + 16*hl + 8];
      v8bf blo = *(const v8bf*)&sWr[(nb+ln)*KP + kb + 16*hl];
      v8bf bhi = *(const v8bf*)&sWr[(nb+ln)*KP + kb + 16*hl + 8];
      if ((kb & 32) == 0)
        acc0 = __builtin_amdgcn_wmma_f32_16x16x32_bf16(false, cat8(alo, ahi),
                                                       false, cat8(blo, bhi),
                                                       (short)0, acc0, false, false);
      else
        acc1 = __builtin_amdgcn_wmma_f32_16x16x32_bf16(false, cat8(alo, ahi),
                                                       false, cat8(blo, bhi),
                                                       (short)0, acc1, false, false);
    }
    // --- gate: wave wv reduces batch row wv: ag = x[wv]·Wrg + ug
    float p = 0.f;
    #pragma unroll
    for (int j = 0; j < 8; ++j) {
      int kk = lane + 32*j;
      p += (float)sXb[wv*KP + kk] * sWrg[kk];
    }
    #pragma unroll
    for (int off = 16; off; off >>= 1) p += __shfl_xor(p, off, 32);
    if (lane == 0) {
      long gi = (long)(bm + wv) * T_ + t;
      float ag = p + AGo[gi];        // AGs region currently holds ug
      AGo[gi] = ag;
      float g = 1.f / (1.f + __expf(-ag));
      Gso[gi] = g;
      sG[wv] = g;
    }
    __syncthreads();   // sG ready; all reads of old sXb complete

    // --- elementwise: a = acc + ui, phi = tanh, x' = g*phi + (1-g)*x
    #pragma unroll
    for (int v = 0; v < 8; ++v) {
      int m = v + 8*hl;                                   // C layout row
      long base = ((long)(bm + m) * T_ + t) * NH_ + nb + ln;
      float a   = acc0[v] + acc1[v]
                + __builtin_nontemporal_load(&Aso[base]); // last use of ui
      __builtin_nontemporal_store(a, &Aso[base]);         // write-once stream
      float phi = fast_tanh(a);
      __builtin_nontemporal_store(phi, &Phio[base]);      // write-once stream
      float g  = sG[m];
      float xn = g * phi + (1.f - g) * xold[v];
      __builtin_nontemporal_store(xn, &Xo[base]);         // write-once stream
      xold[v]  = xn;
      sXb[m*KP + nb + ln] = (__bf16)xn;                   // bf16 mirror for next A
    }
    __syncthreads();   // sXb updated before next step's WMMA reads
  }
}

// ---------------------------------------------------------------------------
// Kernel 4: z = x_last @ Wo^T  (64x64, K=256) — tiny, plain VALU.
// ---------------------------------------------------------------------------
__global__ void __launch_bounds__(256)
zout_kernel(const float* __restrict__ X, const float* __restrict__ Wo,
            float* __restrict__ z) {
  int gid = blockIdx.x * 256 + threadIdx.x;   // 4096 = B*NO
  int b = gid >> 6, o = gid & 63;
  const float* xr = X + ((long)b * T_ + (T_ - 1)) * NH_;
  const float* wr = Wo + (long)o * NH_;
  float s = 0.f;
  for (int h = 0; h < NH_; ++h) s += xr[h] * wr[h];
  z[gid] = s;
}

// ---------------------------------------------------------------------------
extern "C" void kernel_launch(void* const* d_in, const int* in_sizes, int n_in,
                              void* d_out, int out_size, void* d_ws, size_t ws_size,
                              hipStream_t stream) {
  const float* u   = (const float*)d_in[0];
  const float* Wr  = (const float*)d_in[1];
  const float* Wi  = (const float*)d_in[2];
  const float* Wo  = (const float*)d_in[3];
  const float* Wrg = (const float*)d_in[4];
  const float* Wig = (const float*)d_in[5];
  float* out = (float*)d_out;

  const long BT = (long)B_ * T_;
  float* z    = out;                 // [B, NO]
  float* X    = out + (long)B_*NO_;  // [B, T, NH]
  float* As   = X    + BT * NH_;     // [B, T, NH]
  float* Phis = As   + BT * NH_;     // [B, T, NH]
  float* Gs   = Phis + BT * NH_;     // [B, T, 1]
  float* AGs  = Gs   + BT;           // [B, T, 1]

  // 1) ui -> As region, ug -> AGs region (consumed in place by the scan)
  proj_ui_kernel<<<16384, 256, 0, stream>>>(u, Wi, As);
  proj_ug_kernel<<<16384, 256, 0, stream>>>(u, Wig, AGs);

  // 2) persistent scan: 4 WGs x 16 batch rows, ~145 KB LDS each (WGP has 320 KB)
  size_t smem = (size_t)NH_*KP*2 + (size_t)16*KP*2 + (size_t)NH_*4 + 16*4;
  scan_kernel<<<4, 512, smem, stream>>>(Wr, Wrg, X, As, Phis, Gs, AGs);

  // 3) final output projection
  zout_kernel<<<16, 256, 0, stream>>>(X, Wo, z);
}
